// Attention_2362232012982
// MI455X (gfx1250) — matrix operand order repaired and checked by reading, compiled, not run on hardware
//
#include <hip/hip_runtime.h>
#include <cstdint>
#include <cstddef>

// ---------------------------------------------------------------------------
// Types / constants
// ---------------------------------------------------------------------------
typedef __attribute__((ext_vector_type(16))) __bf16 v16bf;
typedef __attribute__((ext_vector_type(8)))  float  v8f;
typedef __attribute__((ext_vector_type(4)))  unsigned int u32x4;
typedef __attribute__((ext_vector_type(8)))  int    i32x8;
typedef __attribute__((ext_vector_type(4)))  int    i32x4;

enum { Bc = 2, Sc = 2048, Dc = 1024, Hc = 16, HDc = 64, Mc = Bc * Sc, Fc = 4 * Dc };

__device__ __forceinline__ __bf16 f2bf(float f) {
  uint32_t u = __builtin_bit_cast(uint32_t, f);
  uint32_t r = u + 0x7FFFu + ((u >> 16) & 1u);   // round-to-nearest-even
  uint16_t h = (uint16_t)(r >> 16);
  return __builtin_bit_cast(__bf16, h);
}

struct FragBits { uint32_t v[8]; };

// A-matrix fragment, 16x32 bf16 (ISA 7.12.2):
//   lanes 0-15  : row M=lane,    K = {0..7, 16..23}
//   lanes 16-31 : row M=lane-16, K = {8..15, 24..31}
__device__ __forceinline__ v16bf load_frag_a(const __bf16* base, int ld) {
  int lane = threadIdx.x & 31;
  int row  = lane & 15;
  int kb   = (lane >> 4) << 3;               // 0 or 8
  const __bf16* p = base + (size_t)row * ld + kb;
  FragBits f;
#pragma unroll
  for (int v = 0; v < 4; ++v) f.v[v]     = *(const uint32_t*)(p + 2 * v);
#pragma unroll
  for (int v = 0; v < 4; ++v) f.v[4 + v] = *(const uint32_t*)(p + 16 + 2 * v);
  return __builtin_bit_cast(v16bf, f);
}

// B-matrix fragment, 32x16 bf16, loaded from TRANSPOSED storage Bt[N][K]:
//   lanes 0-15  : col N=lane,    K = {0..7, 16..23}
//   lanes 16-31 : col N=lane-16, K = {8..15, 24..31}
__device__ __forceinline__ v16bf load_frag_b(const __bf16* base, int ld) {
  int lane = threadIdx.x & 31;
  int col  = lane & 15;
  int kb   = (lane >> 4) << 3;               // 0 or 8
  const __bf16* p = base + (size_t)col * ld + kb;
  FragBits f;
#pragma unroll
  for (int v = 0; v < 4; ++v) f.v[v]     = *(const uint32_t*)(p + 2 * v);
#pragma unroll
  for (int v = 0; v < 4; ++v) f.v[4 + v] = *(const uint32_t*)(p + 16 + 2 * v);
  return __builtin_bit_cast(v16bf, f);
}

__device__ __forceinline__ v8f wmma_bf16(v16bf a, v16bf b, v8f c) {
  return __builtin_amdgcn_wmma_f32_16x16x32_bf16(false, a, false, b, (short)0, c,
                                                 false, false);
}

__device__ __forceinline__ float gelu_exact(float x) {
  return 0.5f * x * (1.0f + erff(x * 0.70710678118654752f));
}

// ---------------------------------------------------------------------------
// Tensor Data Mover: 2D tile DMA (global -> LDS), bf16 elements.
// D# packing per CDNA5 ISA:
//   group0: count=1 | lds_addr | global_addr[56:0] | type=2
//   group1: data_size=1 (2B), tensor dims huge (tiles always in-bounds),
//           tile_dim0=cols, tile_dim1=rows, dim0_stride=stride (elements)
// This toolchain exposes the 6-arg builtin (extra i32x8 descriptor group;
// zero-filled here since the tensor is 2D).
// ---------------------------------------------------------------------------
__device__ __forceinline__ void tdm_load_2d(const void* gsrc, uint32_t lds_addr,
                                            uint32_t rows, uint32_t cols,
                                            uint64_t stride_elems) {
  uint64_t ga = (uint64_t)(uintptr_t)gsrc;
  const uint32_t td = 0x40000000u;           // huge tensor_dim0/1 (no clipping)
  u32x4 g0;
  g0[0] = 1u;                                 // count=1, user descriptor
  g0[1] = lds_addr;                           // LDS byte address
  g0[2] = (uint32_t)ga;                       // global_addr[31:0]
  g0[3] = (uint32_t)((ga >> 32) & 0x01FFFFFFu) | (2u << 30);  // addr msbs, type=2
  i32x8 g1;
  g1[0] = (int)(1u << 16);                    // workgroup_mask=0, data_size=1 (2B)
  g1[1] = (int)((td & 0xFFFFu) << 16);        // tensor_dim0[15:0]
  g1[2] = (int)((td >> 16) | ((td & 0xFFFFu) << 16));   // dim0 msbs | dim1 lsbs
  g1[3] = (int)((td >> 16) | (cols << 16));   // dim1 msbs | tile_dim0
  g1[4] = (int)(rows & 0xFFFFu);              // tile_dim1 | tile_dim2=0
  g1[5] = (int)(uint32_t)stride_elems;        // tensor_dim0_stride[31:0]
  g1[6] = (int)(uint32_t)(stride_elems >> 32);// stride msbs | dim1_stride lsbs=0
  g1[7] = 0;
  i32x4 z4 = {0, 0, 0, 0};                    // groups 2/3 unused (2D tensor)
  i32x8 z8 = {0, 0, 0, 0, 0, 0, 0, 0};
  __builtin_amdgcn_tensor_load_to_lds(g0, g1, z4, z4, z8, 0);
}

// ---------------------------------------------------------------------------
// Conversion kernels
// ---------------------------------------------------------------------------
__global__ __launch_bounds__(256) void conv_bf16(const float* __restrict__ X,
                                                 __bf16* __restrict__ Y, int n) {
  int i = blockIdx.x * 256 + threadIdx.x;
  if (i < n) Y[i] = f2bf(X[i]);
}

// W[K][N] fp32  ->  Wt[N][K] bf16
__global__ __launch_bounds__(256) void conv_transpose_bf16(const float* __restrict__ W,
                                                           __bf16* __restrict__ Wt,
                                                           int K, int N) {
  long long i = (long long)blockIdx.x * 256 + threadIdx.x;
  if (i >= (long long)K * N) return;
  int k = (int)(i / N), n = (int)(i % N);
  Wt[(size_t)n * K + k] = f2bf(W[i]);
}

// ---------------------------------------------------------------------------
// WMMA GEMM, TDM-fed + double-buffered LDS staging.
// C[M,N] = A[M,K](bf16) x Bt[N,K](bf16)^T + bias, fused epilogue:
//   MODE 0: fp32 row-major              MODE 2: bf16 row-major + GELU
//   MODE 3: bf16 head-split [B,H,S,HD]  (Q,K)
//   MODE 4: bf16 head-split-T [B,H,HD,S] (V)
// grid = (M/128, N/64), block = 256 (8 waves; wave -> 32x32 register tile)
// ---------------------------------------------------------------------------
template <int MODE>
__global__ __launch_bounds__(256) void gemm_bf16_wmma(const __bf16* __restrict__ A,
                                                      const __bf16* __restrict__ Bt,
                                                      const float* __restrict__ bias,
                                                      void* __restrict__ Cout,
                                                      int M, int N, int K) {
  __shared__ __bf16 sA[2][128][32];          // 16 KB
  __shared__ __bf16 sB[2][64][32];           // 8 KB
  const int wave = threadIdx.x >> 5;
  const int wm = wave >> 1;                  // 0..3
  const int wn = wave & 1;                   // 0..1
  const int bm0 = blockIdx.x * 128;
  const int bn0 = blockIdx.y * 64;
  const bool issuer = (wave == 0);

  v8f acc[2][2] = {};
  const int NI = K >> 5;

  if (issuer) {                              // prologue: fill buffer 0
    tdm_load_2d(A + (size_t)bm0 * K, (uint32_t)(uintptr_t)&sA[0][0][0], 128, 32, K);
    tdm_load_2d(Bt + (size_t)bn0 * K, (uint32_t)(uintptr_t)&sB[0][0][0], 64, 32, K);
  }

  for (int it = 0; it < NI; ++it) {
    const int cur = it & 1;
    if (issuer) __builtin_amdgcn_s_wait_tensorcnt(0);
    __syncthreads();                         // publish buf[cur]; readers of buf[cur^1] done
    if (issuer && (it + 1 < NI)) {
      const int k0 = (it + 1) << 5;
      tdm_load_2d(A + (size_t)bm0 * K + k0,
                  (uint32_t)(uintptr_t)&sA[cur ^ 1][0][0], 128, 32, K);
      tdm_load_2d(Bt + (size_t)bn0 * K + k0,
                  (uint32_t)(uintptr_t)&sB[cur ^ 1][0][0], 64, 32, K);
    }
    v16bf a0 = load_frag_a(&sA[cur][wm * 32][0], 32);
    v16bf a1 = load_frag_a(&sA[cur][wm * 32 + 16][0], 32);
    v16bf b0 = load_frag_b(&sB[cur][wn * 32][0], 32);
    v16bf b1 = load_frag_b(&sB[cur][wn * 32 + 16][0], 32);
    acc[0][0] = wmma_bf16(a0, b0, acc[0][0]);
    acc[0][1] = wmma_bf16(a0, b1, acc[0][1]);
    acc[1][0] = wmma_bf16(a1, b0, acc[1][0]);
    acc[1][1] = wmma_bf16(a1, b1, acc[1][1]);
  }

  const int m0 = bm0 + wm * 32;
  const int n0 = bn0 + wn * 32;
  const int lane = threadIdx.x & 31;
  const int cn = lane & 15;
  const int rb = (lane >> 4) << 3;           // +8 for hi lanes
#pragma unroll
  for (int im = 0; im < 2; ++im) {
#pragma unroll
    for (int in = 0; in < 2; ++in) {
#pragma unroll
      for (int r = 0; r < 8; ++r) {
        int m = m0 + im * 16 + r + rb;
        int n = n0 + in * 16 + cn;
        float val = acc[im][in][r] + bias[n];
        if (MODE == 0) {
          ((float*)Cout)[(size_t)m * N + n] = val;
        } else if (MODE == 2) {
          ((__bf16*)Cout)[(size_t)m * N + n] = f2bf(gelu_exact(val));
        } else if (MODE == 3) {
          int b = m >> 11, s = m & (Sc - 1);
          int h = n >> 6, hd = n & (HDc - 1);
          size_t idx = (((size_t)(b * Hc + h)) * Sc + s) * HDc + hd;
          ((__bf16*)Cout)[idx] = f2bf(val);
        } else {                             // MODE 4
          int b = m >> 11, s = m & (Sc - 1);
          int h = n >> 6, hd = n & (HDc - 1);
          size_t idx = (((size_t)(b * Hc + h)) * HDc + hd) * Sc + s;
          ((__bf16*)Cout)[idx] = f2bf(val);
        }
      }
    }
  }
}

// ---------------------------------------------------------------------------
// Flash attention: grid = (B*H, S/64), block = 128 (4 waves; wave -> 16 q rows)
// Qh/Kh: [B*H][S][HD] bf16; Vt: [B*H][HD][S] bf16; O: [B*S][D] fp32
// ---------------------------------------------------------------------------
__global__ __launch_bounds__(128) void attn_flash(const __bf16* __restrict__ Qh,
                                                  const __bf16* __restrict__ Kh,
                                                  const __bf16* __restrict__ Vt,
                                                  float* __restrict__ O) {
  __shared__ __bf16 sP[4][16][32];           // per-wave P relayout tile
  const int wv = threadIdx.x >> 5;
  const int lane = threadIdx.x & 31;
  const int cn = lane & 15;
  const int hi = lane >> 4;
  const int bh = blockIdx.x;                 // b*H + h
  const int q0 = blockIdx.y * 64 + wv * 16;

  const __bf16* Qb = Qh + ((size_t)bh * Sc + q0) * HDc;
  v16bf qa0 = load_frag_a(Qb, HDc);          // K dims 0..31
  v16bf qa1 = load_frag_a(Qb + 32, HDc);     // K dims 32..63

  float mrow[8], lrow[8];
#pragma unroll
  for (int r = 0; r < 8; ++r) { mrow[r] = -1e30f; lrow[r] = 0.0f; }
  v8f o0 = {}, o1 = {}, o2 = {}, o3 = {};

  for (int j0 = 0; j0 < Sc; j0 += 32) {
    const __bf16* Kb = Kh + ((size_t)bh * Sc + j0) * HDc;
    // scores: two 16x16 tiles covering keys j0..j0+31
    v8f s1 = {}, s2 = {};
    s1 = wmma_bf16(qa0, load_frag_b(Kb, HDc), s1);
    s1 = wmma_bf16(qa1, load_frag_b(Kb + 32, HDc), s1);
    s2 = wmma_bf16(qa0, load_frag_b(Kb + 16 * HDc, HDc), s2);
    s2 = wmma_bf16(qa1, load_frag_b(Kb + 16 * HDc + 32, HDc), s2);

#pragma unroll
    for (int r = 0; r < 8; ++r) {
      float x1 = s1[r] * 0.125f;             // 1/sqrt(64)
      float x2 = s2[r] * 0.125f;
      float t = fmaxf(x1, x2);
#pragma unroll
      for (int off = 1; off < 16; off <<= 1) t = fmaxf(t, __shfl_xor(t, off, 16));
      float mn = fmaxf(mrow[r], t);
      float al = __expf(mrow[r] - mn);
      float p1 = __expf(x1 - mn);
      float p2 = __expf(x2 - mn);
      float rs = p1 + p2;
#pragma unroll
      for (int off = 1; off < 16; off <<= 1) rs += __shfl_xor(rs, off, 16);
      lrow[r] = lrow[r] * al + rs;
      mrow[r] = mn;
      o0[r] *= al; o1[r] *= al; o2[r] *= al; o3[r] *= al;
      int row = r + (hi << 3);
      sP[wv][row][cn]      = f2bf(p1);       // keys j0..j0+15
      sP[wv][row][16 + cn] = f2bf(p2);       // keys j0+16..j0+31
    }
    asm volatile("s_wait_dscnt 0" ::: "memory");  // LDS store -> load ordering

    v16bf pf = load_frag_a(&sP[wv][0][0], 32);    // P as A-fragment (16x32)
    const __bf16* Vb = Vt + ((size_t)bh * HDc) * Sc + j0;
    o0 = wmma_bf16(pf, load_frag_b(Vb, Sc), o0);
    o1 = wmma_bf16(pf, load_frag_b(Vb + 16 * (size_t)Sc, Sc), o1);
    o2 = wmma_bf16(pf, load_frag_b(Vb + 32 * (size_t)Sc, Sc), o2);
    o3 = wmma_bf16(pf, load_frag_b(Vb + 48 * (size_t)Sc, Sc), o3);
  }

  const int b = bh >> 4, h = bh & 15;
#pragma unroll
  for (int r = 0; r < 8; ++r) {
    float inv = 1.0f / lrow[r];
    size_t gm = (size_t)b * Sc + q0 + r + (hi << 3);
    float* out = O + gm * Dc + h * HDc + cn;
    out[0]  = o0[r] * inv;
    out[16] = o1[r] * inv;
    out[32] = o2[r] * inv;
    out[48] = o3[r] * inv;
  }
}

// ---------------------------------------------------------------------------
// LayerNorm kernels (one block per row of D=1024; 256 threads x 4 elems)
// ---------------------------------------------------------------------------
__global__ __launch_bounds__(256) void ln_residual_to_bf16(
    const float* __restrict__ xin, const float* __restrict__ res,
    const float* __restrict__ g, const float* __restrict__ be,
    __bf16* __restrict__ out) {
  __shared__ float sh[16];
  const int row = blockIdx.x, tid = threadIdx.x;
  const int lane = tid & 31, wv = tid >> 5;
  float v[4], s = 0.f, ss = 0.f;
#pragma unroll
  for (int i = 0; i < 4; ++i) {
    int c = tid + 256 * i;
    v[i] = xin[(size_t)row * Dc + c] + res[(size_t)row * Dc + c];
    s += v[i]; ss += v[i] * v[i];
  }
#pragma unroll
  for (int off = 16; off; off >>= 1) { s += __shfl_xor(s, off, 32); ss += __shfl_xor(ss, off, 32); }
  if (lane == 0) { sh[wv * 2] = s; sh[wv * 2 + 1] = ss; }
  __syncthreads();
  float ts = 0.f, tss = 0.f;
#pragma unroll
  for (int i = 0; i < 8; ++i) { ts += sh[2 * i]; tss += sh[2 * i + 1]; }
  float mu = ts * (1.0f / Dc);
  float rstd = rsqrtf(tss * (1.0f / Dc) - mu * mu + 1e-5f);
#pragma unroll
  for (int i = 0; i < 4; ++i) {
    int c = tid + 256 * i;
    out[(size_t)row * Dc + c] = f2bf((v[i] - mu) * rstd * g[c] + be[c]);
  }
}

__global__ __launch_bounds__(256) void ln_double_to_f32(
    const float* __restrict__ xin, const float* __restrict__ g,
    const float* __restrict__ be, float* __restrict__ out) {
  __shared__ float sh[16];
  const int row = blockIdx.x, tid = threadIdx.x;
  const int lane = tid & 31, wv = tid >> 5;
  float v[4], s = 0.f, ss = 0.f;
#pragma unroll
  for (int i = 0; i < 4; ++i) {
    int c = tid + 256 * i;
    v[i] = 2.0f * xin[(size_t)row * Dc + c];     // ffn + ffn
    s += v[i]; ss += v[i] * v[i];
  }
#pragma unroll
  for (int off = 16; off; off >>= 1) { s += __shfl_xor(s, off, 32); ss += __shfl_xor(ss, off, 32); }
  if (lane == 0) { sh[wv * 2] = s; sh[wv * 2 + 1] = ss; }
  __syncthreads();
  float ts = 0.f, tss = 0.f;
#pragma unroll
  for (int i = 0; i < 8; ++i) { ts += sh[2 * i]; tss += sh[2 * i + 1]; }
  float mu = ts * (1.0f / Dc);
  float rstd = rsqrtf(tss * (1.0f / Dc) - mu * mu + 1e-5f);
#pragma unroll
  for (int i = 0; i < 4; ++i) {
    int c = tid + 256 * i;
    out[(size_t)row * Dc + c] = (v[i] - mu) * rstd * g[c] + be[c];
  }
}

// ---------------------------------------------------------------------------
// Host orchestration
// ---------------------------------------------------------------------------
extern "C" void kernel_launch(void* const* d_in, const int* in_sizes, int n_in,
                              void* d_out, int out_size, void* d_ws, size_t ws_size,
                              hipStream_t stream) {
  const float* x  = (const float*)d_in[0];
  const float* Wq = (const float*)d_in[1];  const float* bq = (const float*)d_in[2];
  const float* Wk = (const float*)d_in[3];  const float* bk = (const float*)d_in[4];
  const float* Wv = (const float*)d_in[5];  const float* bv = (const float*)d_in[6];
  const float* W1 = (const float*)d_in[7];  const float* b1 = (const float*)d_in[8];
  const float* W2 = (const float*)d_in[9];  const float* b2 = (const float*)d_in[10];
  const float* g1 = (const float*)d_in[11]; const float* be1 = (const float*)d_in[12];
  const float* g2 = (const float*)d_in[13]; const float* be2 = (const float*)d_in[14];
  float* out = (float*)d_out;
  (void)in_sizes; (void)n_in; (void)out_size; (void)ws_size;

  char* p = (char*)d_ws;
  auto alloc = [&](size_t bytes) -> char* {
    char* r = p; p += (bytes + 255) & ~(size_t)255; return r;
  };
  __bf16* xb   = (__bf16*)alloc((size_t)Mc * Dc * 2);
  __bf16* Wqt  = (__bf16*)alloc((size_t)Dc * Dc * 2);
  __bf16* Wkt  = (__bf16*)alloc((size_t)Dc * Dc * 2);
  __bf16* Wvt  = (__bf16*)alloc((size_t)Dc * Dc * 2);
  __bf16* W1t  = (__bf16*)alloc((size_t)Dc * Fc * 2);
  __bf16* W2t  = (__bf16*)alloc((size_t)Fc * Dc * 2);
  __bf16* Qhb  = (__bf16*)alloc((size_t)Mc * Dc * 2);
  __bf16* Khb  = (__bf16*)alloc((size_t)Mc * Dc * 2);
  __bf16* Vtb  = (__bf16*)alloc((size_t)Mc * Dc * 2);
  float*  Oat  = (float*) alloc((size_t)Mc * Dc * 4);
  __bf16* h1b  = (__bf16*)alloc((size_t)Mc * Dc * 2);
  __bf16* ffn1 = (__bf16*)alloc((size_t)Mc * Fc * 2);
  float*  f2   = (float*) alloc((size_t)Mc * Dc * 4);

  // 1) fp32 -> bf16 conversions (+ weight transposes)
  conv_bf16<<<(Mc * Dc + 255) / 256, 256, 0, stream>>>(x, xb, Mc * Dc);
  conv_transpose_bf16<<<(Dc * Dc + 255) / 256, 256, 0, stream>>>(Wq, Wqt, Dc, Dc);
  conv_transpose_bf16<<<(Dc * Dc + 255) / 256, 256, 0, stream>>>(Wk, Wkt, Dc, Dc);
  conv_transpose_bf16<<<(Dc * Dc + 255) / 256, 256, 0, stream>>>(Wv, Wvt, Dc, Dc);
  conv_transpose_bf16<<<(Dc * Fc + 255) / 256, 256, 0, stream>>>(W1, W1t, Dc, Fc);
  conv_transpose_bf16<<<(Fc * Dc + 255) / 256, 256, 0, stream>>>(W2, W2t, Fc, Dc);

  // 2) QKV projections (TDM-fed WMMA), head-split epilogues
  dim3 gQKV(Mc / 128, Dc / 64);
  gemm_bf16_wmma<3><<<gQKV, 256, 0, stream>>>(xb, Wqt, bq, Qhb, Mc, Dc, Dc);
  gemm_bf16_wmma<3><<<gQKV, 256, 0, stream>>>(xb, Wkt, bk, Khb, Mc, Dc, Dc);
  gemm_bf16_wmma<4><<<gQKV, 256, 0, stream>>>(xb, Wvt, bv, Vtb, Mc, Dc, Dc);

  // 3) flash attention (WMMA for QK^T and PV)
  attn_flash<<<dim3(Bc * Hc, Sc / 64), 128, 0, stream>>>(Qhb, Khb, Vtb, Oat);

  // 4) residual + LN1 -> bf16
  ln_residual_to_bf16<<<Mc, 256, 0, stream>>>(Oat, x, g1, be1, h1b);

  // 5) FFN GEMM1 (GELU fused) and GEMM2
  gemm_bf16_wmma<2><<<dim3(Mc / 128, Fc / 64), 256, 0, stream>>>(h1b, W1t, b1, ffn1,
                                                                 Mc, Fc, Dc);
  gemm_bf16_wmma<0><<<dim3(Mc / 128, Dc / 64), 256, 0, stream>>>(ffn1, W2t, b2, f2,
                                                                 Mc, Dc, Fc);

  // 6) LN2 of (ffn + ffn) -> d_out
  ln_double_to_f32<<<Mc, 256, 0, stream>>>(f2, g2, be2, out);
}
